// TTQConv2d_42597485641950
// MI455X (gfx1250) — compile-verified, matrix operand order checked by reading, not run on hardware
//
#include <hip/hip_runtime.h>

typedef __attribute__((ext_vector_type(16))) _Float16 v16h;
typedef __attribute__((ext_vector_type(8)))  _Float16 v8h;
typedef __attribute__((ext_vector_type(8)))  float    v8f;

#define B_     64
#define CIN    128
#define HW_IN  56
#define COUT   256
#define HOUT   54
#define P_OUT  (HOUT*HOUT)        // 2916 pixels per image
#define NTOT   (B_*P_OUT)         // 186624 GEMM-N
#define KK     (CIN*9)            // 1152 GEMM-K
#define KSTEPS (KK/32)            // 36
#define LDSSTR 40                 // 32 halfs + 8 pad -> 80B rows (16B aligned)
#define NTILE  256                // block N tile
#define MTILE  128                // block M tile

// ---------------- threshold kernels ----------------
__global__ __launch_bounds__(256) void ttq_wmax_partial(const float* __restrict__ w,
                                                        float* __restrict__ part) {
  __shared__ float sm[256];
  unsigned tid = threadIdx.x;
  unsigned base = blockIdx.x * 1024u + tid;   // 288 * 1024 = 294912 exactly
  float m = 0.f;
#pragma unroll
  for (int i = 0; i < 4; ++i) m = fmaxf(m, fabsf(w[base + i * 256u]));
  sm[tid] = m; __syncthreads();
  for (int s = 128; s > 0; s >>= 1) {
    if (tid < (unsigned)s) sm[tid] = fmaxf(sm[tid], sm[tid + s]);
    __syncthreads();
  }
  if (tid == 0) part[blockIdx.x] = sm[0];
}

__global__ __launch_bounds__(256) void ttq_thresh(const float* __restrict__ part,
                                                  float* __restrict__ tout) {
  __shared__ float sm[256];
  unsigned tid = threadIdx.x;
  float m = part[tid];
  if (tid + 256u < 288u) m = fmaxf(m, part[tid + 256u]);
  sm[tid] = m; __syncthreads();
  for (int s = 128; s > 0; s >>= 1) {
    if (tid < (unsigned)s) sm[tid] = fmaxf(sm[tid], sm[tid + s]);
    __syncthreads();
  }
  if (tid == 0) tout[0] = 0.05f * sm[0];
}

// k -> x element offset table: k = cin*9 + kh*3 + kw
__global__ __launch_bounds__(256) void ttq_offtab(int* __restrict__ offt) {
  for (unsigned k = threadIdx.x; k < KK; k += 256u) {
    unsigned cin = k / 9u, r = k % 9u;
    offt[k] = (int)(cin * (HW_IN * HW_IN) + (r / 3u) * HW_IN + (r % 3u));
  }
}

__device__ __forceinline__ unsigned pack2h(float a, float b) {
  auto h = __builtin_amdgcn_cvt_pkrtz(a, b);  // v_cvt_pk_rz_f16_f32, __fp16x2
  return __builtin_bit_cast(unsigned, h);
}

// ---------------- main implicit-GEMM WMMA conv ----------------
__global__ __launch_bounds__(256) void ttq_conv_wmma(
    const float* __restrict__ x, const float* __restrict__ w,
    const float* __restrict__ bias, const float* __restrict__ wp,
    const float* __restrict__ wn, const float* __restrict__ tptr,
    const int* __restrict__ offt, float* __restrict__ out) {
  __shared__ __align__(16) _Float16 As[MTILE * LDSSTR];  // [cout row][k]
  __shared__ __align__(16) _Float16 Bs[NTILE * LDSSTR];  // [n row][k]

  const unsigned tid     = threadIdx.x;
  const unsigned lane    = tid & 31u;
  const unsigned wvid    = tid >> 5;          // 8 waves
  const unsigned wave_m  = wvid >> 2;         // 0..1  -> 64 rows of M
  const unsigned wave_n  = wvid & 3u;         // 0..3  -> 64 cols of N
  const unsigned lrow    = lane & 15u;
  const unsigned halfsel = lane >> 4;         // ISA 16-bit fragment lane half
  const unsigned kb      = halfsel * 8u;      // K base within fragment

  const unsigned m_base = blockIdx.y * MTILE;
  const unsigned n_base = blockIdx.x * NTILE;

  const float t  = tptr[0];
  const float WP = fabsf(wp[0]);
  const float WN = fabsf(wn[0]);

  // ---- A staging mapping: thread owns (row_t, k_t..k_t+1), +16 rows per jj ----
  const unsigned row_t = tid >> 4;            // 0..15
  const unsigned k_t   = (tid * 2u) & 31u;    // even 0..30
  const float* wbase = w + (size_t)(m_base + row_t) * KK + k_t;

  // ---- B staging mapping: thread owns one n, all 32 k per step ----
  const unsigned nb_off = tid;                // 0..255
  const unsigned n_glob = n_base + nb_off;
  const unsigned b_idx  = n_glob / P_OUT;
  const unsigned p_pix  = n_glob % P_OUT;
  const unsigned oh     = p_pix / HOUT;
  const unsigned ow     = p_pix % HOUT;
  const unsigned xoff   = b_idx * (CIN * HW_IN * HW_IN) + oh * HW_IN + ow;

  unsigned aR[8], bR[16];
  v8f acc[4][4];
#pragma unroll
  for (int i = 0; i < 4; ++i)
#pragma unroll
    for (int j = 0; j < 4; ++j) { v8f z = {}; acc[i][j] = z; }

  auto stage = [&](unsigned s) {
    const unsigned ks = s * 32u;
    // A tile MTILEx32: quantize f32 weights on the fly (weights resident in L2)
#pragma unroll
    for (int jj = 0; jj < 8; ++jj) {
      const float* wpnt = wbase + (size_t)jj * (16u * KK) + ks;
      float w0 = wpnt[0], w1 = wpnt[1];
      float q0 = (w0 > t) ? WP : 0.f;  q0 = (w0 < -t) ? -WN : q0;
      float q1 = (w1 > t) ? WP : 0.f;  q1 = (w1 < -t) ? -WN : q1;
      aR[jj] = pack2h(q0, q1);
    }
    // B tile 32xNTILE: offsets are block-uniform -> SMEM; per-lane xoff in VGPR
#pragma unroll
    for (int jj = 0; jj < 16; ++jj) {
      unsigned kg = ks + (unsigned)jj * 2u;
      int o0 = offt[kg];
      int o1 = offt[kg + 1u];
      float x0 = x[(size_t)o0 + xoff];
      float x1 = x[(size_t)o1 + xoff];
      bR[jj] = pack2h(x0, x1);
    }
  };

  auto commit = [&]() {
#pragma unroll
    for (int jj = 0; jj < 8; ++jj)
      *(unsigned*)&As[(row_t + (unsigned)jj * 16u) * LDSSTR + k_t] = aR[jj];
#pragma unroll
    for (int jj = 0; jj < 16; ++jj)
      *(unsigned*)&Bs[nb_off * LDSSTR + (unsigned)jj * 2u] = bR[jj];
  };

  stage(0);
  commit();
  __syncthreads();

  for (unsigned s = 0; s < KSTEPS; ++s) {
    if (s + 1 < KSTEPS) stage(s + 1);   // overlap global loads with WMMA

    v16h afrag[4], bfrag[4];
#pragma unroll
    for (int i = 0; i < 4; ++i) {
      unsigned row = wave_m * 64u + (unsigned)i * 16u + lrow;
      v8h lo = *(const v8h*)&As[row * LDSSTR + kb];         // K kb..kb+7
      v8h hi = *(const v8h*)&As[row * LDSSTR + kb + 16u];   // K kb+16..kb+23
      afrag[i] = __builtin_shufflevector(lo, hi, 0, 1, 2, 3, 4, 5, 6, 7,
                                         8, 9, 10, 11, 12, 13, 14, 15);
    }
#pragma unroll
    for (int j = 0; j < 4; ++j) {
      unsigned row = wave_n * 64u + (unsigned)j * 16u + lrow;
      v8h lo = *(const v8h*)&Bs[row * LDSSTR + kb];
      v8h hi = *(const v8h*)&Bs[row * LDSSTR + kb + 16u];
      bfrag[j] = __builtin_shufflevector(lo, hi, 0, 1, 2, 3, 4, 5, 6, 7,
                                         8, 9, 10, 11, 12, 13, 14, 15);
    }
#pragma unroll
    for (int i = 0; i < 4; ++i)
#pragma unroll
      for (int j = 0; j < 4; ++j)
        acc[i][j] = __builtin_amdgcn_wmma_f32_16x16x32_f16(
            false, afrag[i], false, bfrag[j], (short)0, acc[i][j], false, false);

    __syncthreads();
    if (s + 1 < KSTEPS) commit();
    __syncthreads();
  }

  // Epilogue: D layout -> VGPR r holds M=r (lanes 0-15) / M=r+8 (lanes 16-31)
#pragma unroll
  for (int j = 0; j < 4; ++j) {
    unsigned nl = n_base + wave_n * 64u + (unsigned)j * 16u + lrow;
    unsigned bb = nl / P_OUT;
    unsigned pp = nl % P_OUT;
    size_t obase = (size_t)bb * (COUT * P_OUT) + pp;
#pragma unroll
    for (int i = 0; i < 4; ++i) {
      unsigned cbase = m_base + wave_m * 64u + (unsigned)i * 16u + halfsel * 8u;
#pragma unroll
      for (int r = 0; r < 8; ++r) {
        unsigned c = cbase + r;
        out[obase + (size_t)c * P_OUT] = acc[i][j][r] + bias[c];
      }
    }
  }
}

extern "C" void kernel_launch(void* const* d_in, const int* in_sizes, int n_in,
                              void* d_out, int out_size, void* d_ws, size_t ws_size,
                              hipStream_t stream) {
  const float* x    = (const float*)d_in[0];
  const float* w    = (const float*)d_in[1];
  const float* bias = (const float*)d_in[2];
  const float* wp   = (const float*)d_in[3];
  const float* wn   = (const float*)d_in[4];
  float* out = (float*)d_out;

  float* wsf  = (float*)d_ws;
  float* part = wsf;                 // 288 partial maxima
  float* tptr = wsf + 512;           // threshold scalar
  int*   offt = (int*)(wsf + 1024);  // 1152 k->offset entries

  ttq_offtab<<<1, 256, 0, stream>>>(offt);
  ttq_wmax_partial<<<288, 256, 0, stream>>>(w, part);
  ttq_thresh<<<1, 256, 0, stream>>>(part, tptr);

  dim3 grid(NTOT / NTILE, COUT / MTILE);   // 729 x 2
  ttq_conv_wmma<<<grid, 256, 0, stream>>>(x, w, bias, wp, wn, tptr, offt, out);
}
